// SimpleLinearAttention_14551349199460
// MI455X (gfx1250) — matrix operand order repaired and checked by reading, not compiled, not run on hardware
//
#include <hip/hip_runtime.h>
#include <hip/hip_bf16.h>
#include <stdint.h>

// SimpleLinearAttention for MI455X (gfx1250), bf16 WMMA pipeline.
// B=16, C=256, N=4096, heads=8, hd=32.

#define EPSF 1e-6f

typedef __bf16 v16bf __attribute__((ext_vector_type(16)));
typedef float  v8f   __attribute__((ext_vector_type(8)));
typedef unsigned int u32x4 __attribute__((ext_vector_type(4)));
typedef float f32x4 __attribute__((ext_vector_type(4)));

union Frag { v16bf v; u32x4 q[2]; unsigned short s[16]; };

__device__ __forceinline__ unsigned short f2bf(float f) {
  unsigned u = __float_as_uint(f);
  u += 0x7FFFu + ((u >> 16) & 1u);          // round-to-nearest-even
  return (unsigned short)(u >> 16);
}
__device__ __forceinline__ float bf2f(unsigned short h) {
  return __uint_as_float(((unsigned)h) << 16);
}

__device__ __forceinline__ void wait_asynccnt0() {
#if __has_builtin(__builtin_amdgcn_s_wait_asynccnt)
  __builtin_amdgcn_s_wait_asynccnt(0);
#else
  asm volatile("s_wait_asynccnt 0x0" ::: "memory");
#endif
}

// Async memory->LDS copy of 8 bytes per active lane (GLOBAL_LOAD_ASYNC_TO_LDS_B64,
// tracked by ASYNCcnt).  lds_off is the LDS byte offset (low 32 bits of the
// generic shared pointer), gsrc the per-lane global address.
__device__ __forceinline__ void async_copy_b64(unsigned lds_off, const void* gsrc) {
  asm volatile("global_load_async_to_lds_b64 %0, %1, off"
               :: "v"(lds_off), "v"(gsrc) : "memory");
}

// ---------------------------------------------------------------------------
// K1: qkv = x^T @ W_qkv^T + b_qkv  (per batch).  A is K-major in memory
// (x is (C,N)), so stage transposed into LDS as lA[m][k].  Output bf16 into
// separate q/k/v buffers, with +EPS fused for q,k.
// grid: (N/64, B, 768/64), block 256 (8 waves: 4 m-subtiles x 2 j-groups)
// ---------------------------------------------------------------------------
__global__ __launch_bounds__(256) void k1_qkv(
    const float* __restrict__ x, const float* __restrict__ Wqkv,
    const float* __restrict__ bqkv,
    unsigned short* __restrict__ qb, unsigned short* __restrict__ kb,
    unsigned short* __restrict__ vb) {
  __shared__ unsigned short lA[64][32];
  __shared__ unsigned short lB[64][32];
  const int tid = threadIdx.x, lane = tid & 31, w = tid >> 5;
  const int wm = w & 3, wn = w >> 2;
  const int b = blockIdx.y, m0 = blockIdx.x * 64, jblk = blockIdx.z;
  const int jbase = jblk * 64;
  unsigned short* outp = (jblk < 4) ? qb : (jblk < 8) ? kb : vb;
  const float eps = (jblk < 8) ? EPSF : 0.0f;
  const int colbuf = (jblk & 3) * 64;

  v8f acc0 = {}; v8f acc1 = {};
  const int akk = tid >> 3, amm = (tid & 7) * 8;      // A stage: (k,m) read
  const int bj  = tid >> 2, bkw = (tid & 3) * 8;      // B stage: (j,k) read
  const int kb2 = (lane >> 4) * 8, kb3 = (lane >> 4) * 16;
  const int mrow = wm * 16 + (lane & 15);
  const int jl0 = wn * 32 + (lane & 15);
  const int jl1 = jl0 + 16;

  for (int kt = 0; kt < 8; ++kt) {
    const int k0 = kt * 32;
    // stage A tile: x[b][k0+akk][m0+amm .. +7]  -> lA[m][k] (transposed)
    const float* ax = x + ((size_t)(b * 256 + k0 + akk)) * 4096 + m0 + amm;
    const float* bw = Wqkv + (size_t)(jbase + bj) * 256 + k0 + bkw;
    if (kt < 7) {   // prefetch next K-tile (lowers to global_prefetch_b8)
      __builtin_prefetch(ax + 32 * 4096, 0, 3);
      __builtin_prefetch(bw + 32, 0, 3);
    }
    f32x4 a0 = *(const f32x4*)ax;
    f32x4 a1 = *(const f32x4*)(ax + 4);
#pragma unroll
    for (int i = 0; i < 4; ++i) lA[amm + i][akk] = f2bf(a0[i]);
#pragma unroll
    for (int i = 0; i < 4; ++i) lA[amm + 4 + i][akk] = f2bf(a1[i]);
    // stage B tile: W_qkv[jbase+bj][k0+bkw .. +7] -> lB[j][k]
    f32x4 b0 = *(const f32x4*)bw;
    f32x4 b1 = *(const f32x4*)(bw + 4);
#pragma unroll
    for (int i = 0; i < 4; ++i) lB[bj][bkw + i] = f2bf(b0[i]);
#pragma unroll
    for (int i = 0; i < 4; ++i) lB[bj][bkw + 4 + i] = f2bf(b1[i]);
    __syncthreads();

    Frag a, fb0, fb1;
    a.q[0]   = *(const u32x4*)&lA[mrow][kb2];
    a.q[1]   = *(const u32x4*)&lA[mrow][kb2 + 16];
    fb0.q[0] = *(const u32x4*)&lB[jl0][kb2];
    fb0.q[1] = *(const u32x4*)&lB[jl0][kb2 + 16];
    fb1.q[0] = *(const u32x4*)&lB[jl1][kb2];
    fb1.q[1] = *(const u32x4*)&lB[jl1][kb2 + 16];
    acc0 = __builtin_amdgcn_wmma_f32_16x16x32_bf16(false, a.v, false, fb0.v,
                                                   (short)0, acc0, false, false);
    acc1 = __builtin_amdgcn_wmma_f32_16x16x32_bf16(false, a.v, false, fb1.v,
                                                   (short)0, acc1, false, false);
    __syncthreads();
  }
#pragma unroll
  for (int nt = 0; nt < 2; ++nt) {
    const v8f acc = nt ? acc1 : acc0;
    const int coll = wn * 32 + nt * 16 + (lane & 15);
    const float bias = bqkv[jbase + coll] + eps;
    const int col = colbuf + coll;
#pragma unroll
    for (int r = 0; r < 8; ++r) {
      const int m = wm * 16 + (lane >> 4) * 8 + r;   // C/D layout: M = r + 8*(lane>=16)
      outp[((size_t)(b * 4096 + m0 + m)) * 256 + col] = f2bf(acc[r] + bias);
    }
  }
}

// ---------------------------------------------------------------------------
// K2: L2-normalize q and k rows (full C=256) in place. One wave per row.
// grid: 65536 rows / 8 waves = 8192 blocks
// ---------------------------------------------------------------------------
__global__ __launch_bounds__(256) void k2_norm(unsigned short* __restrict__ qb,
                                               unsigned short* __restrict__ kb) {
  const int lane = threadIdx.x & 31, w = threadIdx.x >> 5;
  const size_t row = (size_t)blockIdx.x * 8 + w;
#pragma unroll
  for (int pass = 0; pass < 2; ++pass) {
    unsigned short* p = (pass ? kb : qb) + row * 256 + lane * 8;
    u32x4 d = *(u32x4*)p;
    float f[8];
#pragma unroll
    for (int i = 0; i < 4; ++i) {
      f[2 * i]     = bf2f((unsigned short)(d[i] & 0xffffu));
      f[2 * i + 1] = bf2f((unsigned short)(d[i] >> 16));
    }
    float s = 0.f;
#pragma unroll
    for (int i = 0; i < 8; ++i) s += f[i] * f[i];
#pragma unroll
    for (int off = 16; off > 0; off >>= 1) s += __shfl_xor(s, off, 32);
    const float inv = rsqrtf(s);
    u32x4 o;
#pragma unroll
    for (int i = 0; i < 4; ++i) {
      unsigned lo = f2bf(f[2 * i] * inv), hi = f2bf(f[2 * i + 1] * inv);
      o[i] = lo | (hi << 16);
    }
    *(u32x4*)p = o;
  }
}

// ---------------------------------------------------------------------------
// K3: kv[bh] = k^T @ v (32x32, reduce over N=4096) and ksum[bh] = sum_n k.
// One block per bh; each wave reduces its own n-range with 2x2 WMMA tiles,
// then fp32 cross-wave reduction in LDS.  kv is written TRANSPOSED ([d][c])
// and in bf16 so K4 can stage it with a raw async memory->LDS copy (the bf16
// rounding is identical to what K4 previously applied at staging time).
// ---------------------------------------------------------------------------
__global__ __launch_bounds__(256) void k3_kv(
    const unsigned short* __restrict__ kb, const unsigned short* __restrict__ vb,
    unsigned short* __restrict__ kvTout, float* __restrict__ ksout) {
  __shared__ unsigned short stg[8][2][32][32];   // per-wave (k,v) tiles, 32 KiB
  __shared__ float ksred[8][32];
  const int tid = threadIdx.x, lane = tid & 31, w = tid >> 5;
  const int bh = blockIdx.x, b = bh >> 3, hc = (bh & 7) * 32;
  const size_t base = ((size_t)b * 4096) * 256 + hc;
  v8f acc[2][2] = {};
  float ks = 0.f;
  const int kb2 = (lane >> 4) * 8, kb3 = (lane >> 4) * 16;
  for (int it = 0; it < 16; ++it) {
    const int n0 = w * 512 + it * 32;
    const unsigned short* kr = kb + base + (size_t)(n0 + lane) * 256;
    const unsigned short* vr = vb + base + (size_t)(n0 + lane) * 256;
#pragma unroll
    for (int c4 = 0; c4 < 4; ++c4) {
      u32x4 dk = *(const u32x4*)(kr + c4 * 8);
      u32x4 dv = *(const u32x4*)(vr + c4 * 8);
#pragma unroll
      for (int i = 0; i < 4; ++i) {
        stg[w][0][c4 * 8 + 2 * i][lane]     = (unsigned short)(dk[i] & 0xffffu);
        stg[w][0][c4 * 8 + 2 * i + 1][lane] = (unsigned short)(dk[i] >> 16);
        stg[w][1][c4 * 8 + 2 * i][lane]     = (unsigned short)(dv[i] & 0xffffu);
        stg[w][1][c4 * 8 + 2 * i + 1][lane] = (unsigned short)(dv[i] >> 16);
      }
    }
    __syncthreads();
#pragma unroll
    for (int nn = 0; nn < 32; ++nn) ks += bf2f(stg[w][0][lane][nn]);
    Frag fa[2], fbv[2];
#pragma unroll
    for (int mt = 0; mt < 2; ++mt) {
      const int c = mt * 16 + (lane & 15);
      fa[mt].q[0] = *(const u32x4*)&stg[w][0][c][kb2];
      fa[mt].q[1] = *(const u32x4*)&stg[w][0][c][kb2 + 16];
    }
#pragma unroll
    for (int nt = 0; nt < 2; ++nt) {
      const int dd = nt * 16 + (lane & 15);
      fbv[nt].q[0] = *(const u32x4*)&stg[w][1][dd][kb2];
      fbv[nt].q[1] = *(const u32x4*)&stg[w][1][dd][kb2 + 16];
    }
#pragma unroll
    for (int mt = 0; mt < 2; ++mt)
#pragma unroll
      for (int nt = 0; nt < 2; ++nt)
        acc[mt][nt] = __builtin_amdgcn_wmma_f32_16x16x32_bf16(
            false, fa[mt].v, false, fbv[nt].v, (short)0, acc[mt][nt], false, false);
    __syncthreads();
  }
  float* red = (float*)stg;   // reuse staging LDS: 8 x 1024 fp32
#pragma unroll
  for (int mt = 0; mt < 2; ++mt)
#pragma unroll
    for (int nt = 0; nt < 2; ++nt)
#pragma unroll
      for (int r = 0; r < 8; ++r) {
        const int c  = mt * 16 + (lane >> 4) * 8 + r;
        const int dd = nt * 16 + (lane & 15);
        red[w * 1024 + c * 32 + dd] = acc[mt][nt][r];
      }
  ksred[w][lane] = ks;
  __syncthreads();
  for (int p = tid; p < 1024; p += 256) {
    float s = 0.f;
#pragma unroll
    for (int ww = 0; ww < 8; ++ww) s += red[ww * 1024 + p];
    // p = c*32 + d ; store transposed as [d][c] in bf16
    kvTout[(size_t)bh * 1024 + (p & 31) * 32 + (p >> 5)] = f2bf(s);
  }
  if (tid < 32) {
    float s = 0.f;
#pragma unroll
    for (int ww = 0; ww < 8; ++ww) s += ksred[ww][tid];
    ksout[bh * 32 + tid] = s;
  }
}

// ---------------------------------------------------------------------------
// K4: attn = (q @ kv) * z, heads merged into (B,N,256) bf16.
// kv tile (pre-transposed bf16, 2 KiB) and ksum (128 B) are staged into LDS
// with GLOBAL_LOAD_ASYNC_TO_LDS_B64 (ASYNCcnt path); q A-fragments come
// straight from global (K is contiguous); z = 1/(q . ksum + eps) per row.
// grid: (N/64, BH), block 256 (8 waves -> 4x2 16x16 tiles)
// ---------------------------------------------------------------------------
__global__ __launch_bounds__(256) void k4_attn(
    const unsigned short* __restrict__ qb, const unsigned short* __restrict__ kvT_g,
    const float* __restrict__ ksum, unsigned short* __restrict__ attn) {
  __shared__ unsigned short kvT[32][32];   // [d][c]
  __shared__ float ksl[32];
  __shared__ float zl[64];
  const int tid = threadIdx.x, lane = tid & 31, w = tid >> 5;
  const int bh = blockIdx.y, b = bh >> 3, hc = (bh & 7) * 32;
  const int n0 = blockIdx.x * 64;
  // async memory->LDS staging: 256 lanes x 8 B = 2 KiB kv tile
  {
    const unsigned lds_kv = (unsigned)(size_t)(&kvT[0][0]) + (unsigned)tid * 8u;
    async_copy_b64(lds_kv, kvT_g + (size_t)bh * 1024 + tid * 4);
    if (tid < 16) {
      const unsigned lds_ks = (unsigned)(size_t)(&ksl[0]) + (unsigned)tid * 8u;
      async_copy_b64(lds_ks, ksum + bh * 32 + tid * 2);
    }
    wait_asynccnt0();   // each wave drains its own ASYNCcnt ...
  }
  __syncthreads();      // ... then the barrier publishes LDS to all waves
  if (tid < 64) {
    const unsigned short* qr = qb + ((size_t)(b * 4096 + n0 + tid)) * 256 + hc;
    float s = 0.f;
#pragma unroll
    for (int c = 0; c < 32; ++c) s += bf2f(qr[c]) * ksl[c];
    zl[tid] = 1.f / (s + EPSF);
  }
  __syncthreads();
  const int mt = w >> 1, nt = w & 1;
  const int kb2 = (lane >> 4) * 8, kb3 = (lane >> 4) * 16;
  const int m = mt * 16 + (lane & 15);
  const unsigned short* qr = qb + ((size_t)(b * 4096 + n0 + m)) * 256 + hc;
  Frag a, fb;
  a.q[0] = *(const u32x4*)(qr + kb2);
  a.q[1] = *(const u32x4*)(qr + kb2 + 16);
  const int dd = nt * 16 + (lane & 15);
  fb.q[0] = *(const u32x4*)&kvT[dd][kb2];
  fb.q[1] = *(const u32x4*)&kvT[dd][kb2 + 16];
  v8f acc = {};
  acc = __builtin_amdgcn_wmma_f32_16x16x32_bf16(false, a.v, false, fb.v,
                                                (short)0, acc, false, false);
#pragma unroll
  for (int r = 0; r < 8; ++r) {
    const int mm = mt * 16 + (lane >> 4) * 8 + r;
    attn[((size_t)(b * 4096 + n0 + mm)) * 256 + hc + dd] = f2bf(acc[r] * zl[mm]);
  }
}

// ---------------------------------------------------------------------------
// K5: out = attn @ W_proj^T + b_proj, stored transposed as (B, C, N).
// The WMMA D layout gives each lane 8 consecutive M(=n) values for a fixed
// column(=channel), so the transposed store is two coalesced float4 stores.
// grid: (N/64, B, 256/64), block 256 (2 wm x 4 wn waves)
// ---------------------------------------------------------------------------
__global__ __launch_bounds__(256) void k5_proj(
    const unsigned short* __restrict__ attn, const float* __restrict__ Wp,
    const float* __restrict__ bp, float* __restrict__ out) {
  __shared__ unsigned short lB[64][32];
  const int tid = threadIdx.x, lane = tid & 31, w = tid >> 5;
  const int wm = w >> 2, wn = w & 3;
  const int b = blockIdx.y, m0 = blockIdx.x * 64, jbase = blockIdx.z * 64;
  v8f acc[2] = {};
  const int bj = tid >> 2, bkw = (tid & 3) * 8;
  const int kb2 = (lane >> 4) * 8, kb3 = (lane >> 4) * 16;
  const int jl = wn * 16 + (lane & 15);
  for (int kt = 0; kt < 8; ++kt) {
    const int k0 = kt * 32;
    const float* bw = Wp + (size_t)(jbase + bj) * 256 + k0 + bkw;
    if (kt < 7) __builtin_prefetch(bw + 32, 0, 3);
    f32x4 b0 = *(const f32x4*)bw;
    f32x4 b1 = *(const f32x4*)(bw + 4);
#pragma unroll
    for (int i = 0; i < 4; ++i) lB[bj][bkw + i] = f2bf(b0[i]);
#pragma unroll
    for (int i = 0; i < 4; ++i) lB[bj][bkw + 4 + i] = f2bf(b1[i]);
    __syncthreads();
    Frag fb;
    fb.q[0] = *(const u32x4*)&lB[jl][kb2];
    fb.q[1] = *(const u32x4*)&lB[jl][kb2 + 16];
#pragma unroll
    for (int mt = 0; mt < 2; ++mt) {
      const int m = wm * 32 + mt * 16 + (lane & 15);
      const unsigned short* ar = attn + ((size_t)(b * 4096 + m0 + m)) * 256 + k0;
      Frag a;
      a.q[0] = *(const u32x4*)(ar + kb2);
      a.q[1] = *(const u32x4*)(ar + kb2 + 16);
      acc[mt] = __builtin_amdgcn_wmma_f32_16x16x32_bf16(false, a.v, false, fb.v,
                                                        (short)0, acc[mt], false, false);
    }
    __syncthreads();
  }
  const int cout = jbase + jl;
  const float bias = bp[cout];
#pragma unroll
  for (int mt = 0; mt < 2; ++mt) {
    const int nb = m0 + wm * 32 + mt * 16 + (lane >> 4) * 8;
    float* op = out + ((size_t)(b * 256 + cout)) * 4096 + nb;
    f32x4 o0, o1;
#pragma unroll
    for (int r = 0; r < 4; ++r) { o0[r] = acc[mt][r] + bias; o1[r] = acc[mt][r + 4] + bias; }
    *(f32x4*)op = o0;
    *(f32x4*)(op + 4) = o1;
  }
}

// ---------------------------------------------------------------------------
extern "C" void kernel_launch(void* const* d_in, const int* in_sizes, int n_in,
                              void* d_out, int out_size, void* d_ws, size_t ws_size,
                              hipStream_t stream) {
  (void)in_sizes; (void)n_in; (void)out_size; (void)ws_size;
  const float* x    = (const float*)d_in[0];
  const float* Wqkv = (const float*)d_in[1];
  const float* bqkv = (const float*)d_in[2];
  const float* Wp   = (const float*)d_in[3];
  const float* bp   = (const float*)d_in[4];
  float* out = (float*)d_out;
  char* ws = (char*)d_ws;
  const size_t SZ = (size_t)16 * 4096 * 256 * sizeof(unsigned short);  // 32 MiB
  unsigned short* qb   = (unsigned short*)(ws);
  unsigned short* kbf  = (unsigned short*)(ws + SZ);
  unsigned short* vb   = (unsigned short*)(ws + 2 * SZ);
  unsigned short* att  = (unsigned short*)(ws + 3 * SZ);
  unsigned short* kvbf = (unsigned short*)(ws + 4 * SZ);                 // 128*1024 bf16
  float* ksum = (float*)(ws + 4 * SZ + (size_t)128 * 1024 * sizeof(unsigned short));

  hipLaunchKernelGGL(k1_qkv,  dim3(64, 16, 12), dim3(256), 0, stream,
                     x, Wqkv, bqkv, qb, kbf, vb);
  hipLaunchKernelGGL(k2_norm, dim3(8192),       dim3(256), 0, stream, qb, kbf);
  hipLaunchKernelGGL(k3_kv,   dim3(128),        dim3(256), 0, stream,
                     kbf, vb, kvbf, ksum);
  hipLaunchKernelGGL(k4_attn, dim3(64, 128),    dim3(256), 0, stream,
                     qb, kvbf, ksum, att);
  hipLaunchKernelGGL(k5_proj, dim3(64, 16, 4),  dim3(256), 0, stream,
                     att, Wp, bp, out);
}
